// GraphConvolution_77799037599872
// MI455X (gfx1250) — matrix operand order repaired and checked by reading, not compiled, not run on hardware
//
#include <hip/hip_runtime.h>
#include <hip/hip_bf16.h>

// ---------------------------------------------------------------------------
// CDNA5 / gfx1250 (MI455X) implementation. wave32, WMMA 16x16x32 bf16.
// ---------------------------------------------------------------------------

typedef __attribute__((ext_vector_type(16))) __bf16 bf16x16;
typedef __attribute__((ext_vector_type(8)))  float  floatx8;

#define WMMA_BF16(A_, B_, C_) \
  __builtin_amdgcn_wmma_f32_16x16x32_bf16(false, (A_), false, (B_), (short)0, (C_), false, false)

// --- WMMA fragment loaders (contiguous k order k = 16h + s, A and B alike) -------------------
// A 16x32 bf16: lane L -> row = L&15, half h = L>>4; VGPR j holds K pair at
//   k0 = h*16 + 2*j.   (src row-major, K contiguous, ld elems)
__device__ __forceinline__ bf16x16 load_frag_a(const __bf16* src, int ld, int lane) {
  bf16x16 a;
  const int row = lane & 15, hh = lane >> 4;
#pragma unroll
  for (int j = 0; j < 8; ++j) {
    const int k = (hh << 4) + (j << 1);
    a[2 * j]     = src[row * ld + k];
    a[2 * j + 1] = src[row * ld + k + 1];
  }
  return a;
}
// B 32x16 bf16: lane L -> col = L&15, half h = L>>4; VGPR j holds K pair at
//   k0 = h*16 + 2*j.  Two variants:
//   _T:       source stored transposed [col][k]  (K contiguous)
//   _strided: source stored [k][col]             (K strided by ld)
__device__ __forceinline__ bf16x16 load_frag_b_T(const __bf16* srcT, int ld, int lane) {
  bf16x16 b;
  const int col = lane & 15, hh = lane >> 4;
#pragma unroll
  for (int j = 0; j < 8; ++j) {
    const int k = (hh << 4) + (j << 1);
    b[2 * j]     = srcT[col * ld + k];
    b[2 * j + 1] = srcT[col * ld + k + 1];
  }
  return b;
}
__device__ __forceinline__ bf16x16 load_frag_b_strided(const __bf16* src, int ld, int lane) {
  bf16x16 b;
  const int col = lane & 15, hh = lane >> 4;
#pragma unroll
  for (int j = 0; j < 8; ++j) {
    const int k = (hh << 4) + (j << 1);
    b[2 * j]     = src[k * ld + col];
    b[2 * j + 1] = src[(k + 1) * ld + col];
  }
  return b;
}

// ---------------------------------------------------------------------------
// Kernel 1: per (b,h) fused  O1 = W1 X_l + b1, O2 = W2 X_l + b2,
//           ATT += O2^T O1  accumulated over l, scaled by 1/(o*l).
// grid (256,3), block 128 (4 waves).
// ---------------------------------------------------------------------------
__global__ __launch_bounds__(128) void k_attm(
    const float* __restrict__ x, const float* __restrict__ cw1, const float* __restrict__ cb1,
    const float* __restrict__ cw2, const float* __restrict__ cb2, float* __restrict__ att_out)
{
  __shared__ __bf16 W1s[64 * 64];   // [o][c]
  __shared__ __bf16 W2s[64 * 64];   // [o][c]
  __shared__ __bf16 XsT[32 * 64];   // [n][c]
  __shared__ __bf16 O1T[32 * 64];   // [n][o]
  __shared__ __bf16 O2T[32 * 64];   // [m][o]
  __shared__ float  red[4 * 256];   // cross-wave K reduction

  const int b = blockIdx.x, h = blockIdx.y;
  const int tid = threadIdx.x, w = tid >> 5, lane = tid & 31;
  const int hh = lane >> 4, col = lane & 15;

  const float* w1g = cw1 + h * 64 * 64;
  const float* w2g = cw2 + h * 64 * 64;
  for (int i = tid; i < 64 * 64; i += 128) {
    W1s[i] = (__bf16)w1g[i];
    W2s[i] = (__bf16)w2g[i];
  }

  float bias1[8], bias2[8];
#pragma unroll
  for (int r = 0; r < 8; ++r) {
    const int o = 16 * w + r + 8 * hh;
    bias1[r] = cb1[h * 64 + o];
    bias2[r] = cb2[h * 64 + o];
  }

  const int ks = w & 1;   // K-slice of the attention contraction (o dim)
  const int mt = w >> 1;  // M-tile of ATT
  floatx8 att0 = {}, att1 = {};

  __syncthreads();

  const float* xb = x + (size_t)b * 64 * 23 * 35;
  for (int l = 0; l < 35; ++l) {
    // stage X_l^T : XsT[n][c] = x[b,c,n,l]
    for (int i = tid; i < 23 * 64; i += 128) {
      const int n = i >> 6, c = i & 63;
      XsT[n * 64 + c] = (__bf16)xb[(c * 23 + n) * 35 + l];
      if (l + 1 < 35) __builtin_prefetch(&xb[(c * 23 + n) * 35 + l + 1], 0, 0);
    }
    __syncthreads();

    // projection GEMMs: wave w owns output rows o in [16w, 16w+16)
    floatx8 acc1[2], acc2[2];
#pragma unroll
    for (int nt = 0; nt < 2; ++nt)
#pragma unroll
      for (int r = 0; r < 8; ++r) { acc1[nt][r] = bias1[r]; acc2[nt][r] = bias2[r]; }

#pragma unroll
    for (int kk = 0; kk < 2; ++kk) {
      const bf16x16 fa1 = load_frag_a(W1s + (16 * w) * 64 + kk * 32, 64, lane);
      const bf16x16 fa2 = load_frag_a(W2s + (16 * w) * 64 + kk * 32, 64, lane);
#pragma unroll
      for (int nt = 0; nt < 2; ++nt) {
        const bf16x16 fb = load_frag_b_T(XsT + (nt * 16) * 64 + kk * 32, 64, lane);
        acc1[nt] = WMMA_BF16(fa1, fb, acc1[nt]);
        acc2[nt] = WMMA_BF16(fa2, fb, acc2[nt]);
      }
    }
    // write transposed bf16 tiles: O1T[n][o], O2T[m][o]
#pragma unroll
    for (int nt = 0; nt < 2; ++nt)
#pragma unroll
      for (int r = 0; r < 8; ++r) {
        const int o = 16 * w + r + 8 * hh, n = nt * 16 + col;
        O1T[n * 64 + o] = (__bf16)acc1[nt][r];
        O2T[n * 64 + o] = (__bf16)acc2[nt][r];
      }
    __syncthreads();

    // ATT partial: rows m in [16*mt,..), K slice ks (o in [32*ks, 32*ks+32))
    const bf16x16 fam = load_frag_a(O2T + (mt * 16) * 64 + ks * 32, 64, lane);
    const bf16x16 fb0 = load_frag_b_T(O1T + 0 * 64 + ks * 32, 64, lane);
    const bf16x16 fb1 = load_frag_b_T(O1T + 16 * 64 + ks * 32, 64, lane);
    att0 = WMMA_BF16(fam, fb0, att0);
    att1 = WMMA_BF16(fam, fb1, att1);
    // next iteration's XsT staging barrier also orders these LDS reads
  }
  __syncthreads();

  if (ks == 1) {
#pragma unroll
    for (int r = 0; r < 8; ++r) {
      red[(mt * 2 + 0) * 256 + (r + 8 * hh) * 16 + col] = att0[r];
      red[(mt * 2 + 1) * 256 + (r + 8 * hh) * 16 + col] = att1[r];
    }
  }
  __syncthreads();
  if (ks == 0) {
    const float inv = 1.0f / 2240.0f;  // 1/(o*l)
#pragma unroll
    for (int r = 0; r < 8; ++r) {
      const int m = mt * 16 + r + 8 * hh;
      const float v0 = (att0[r] + red[(mt * 2 + 0) * 256 + (r + 8 * hh) * 16 + col]) * inv;
      const float v1 = (att1[r] + red[(mt * 2 + 1) * 256 + (r + 8 * hh) * 16 + col]) * inv;
      if (m < 23) {
        if (col < 23)      att_out[((b * 3 + h) * 23 + m) * 23 + col] = v0;
        if (16 + col < 23) att_out[((b * 3 + h) * 23 + m) * 23 + 16 + col] = v1;
      }
    }
  }
}

// ---------------------------------------------------------------------------
// Kernel 2: BatchNorm batch statistics over b=256, folded to scale/shift.
// grid (529,3), block 256 (thread == batch index).
// ---------------------------------------------------------------------------
__global__ __launch_bounds__(256) void k_bnstats(
    const float* __restrict__ att, const float* __restrict__ gamma,
    const float* __restrict__ beta, float* __restrict__ scale, float* __restrict__ shift)
{
  __shared__ float s[256], sq[256];
  const int f = blockIdx.x, h = blockIdx.y, t = threadIdx.x;
  const float v = att[(t * 3 + h) * 529 + f];
  s[t] = v; sq[t] = v * v;
  __syncthreads();
  for (int st = 128; st > 0; st >>= 1) {
    if (t < st) { s[t] += s[t + st]; sq[t] += sq[t + st]; }
    __syncthreads();
  }
  if (t == 0) {
    const float mean = s[0] * (1.0f / 256.0f);
    const float var  = sq[0] * (1.0f / 256.0f) - mean * mean;
    const float sc   = gamma[h * 529 + f] * rsqrtf(var + 1e-5f);
    scale[h * 529 + f] = sc;
    shift[h * 529 + f] = beta[h * 529 + f] - mean * sc;
  }
}

// ---------------------------------------------------------------------------
// Kernel 3: normalize + softmax(axis=-2) + A_ske + att  ->  zero-padded
// bf16 A [b][h][32(k=n)][32(m)].  grid (256,3), block 128.
// ---------------------------------------------------------------------------
__global__ __launch_bounds__(128) void k_softmax_A(
    const float* __restrict__ att, const float* __restrict__ scale,
    const float* __restrict__ shift, const float* __restrict__ attp,
    const float* __restrict__ A_ske, __bf16* __restrict__ A_bf)
{
  __shared__ float F[529];
  __shared__ float Af[529];
  const int b = blockIdx.x, h = blockIdx.y, t = threadIdx.x;
  for (int i = t; i < 529; i += 128)
    F[i] = att[(b * 3 + h) * 529 + i] * scale[h * 529 + i] + shift[h * 529 + i];
  __syncthreads();
  if (t < 23) {
    const int j = t;  // column; softmax over row index i (axis -2)
    float mx = -1e30f;
    for (int i = 0; i < 23; ++i) mx = fmaxf(mx, F[i * 23 + j]);
    float sum = 0.0f;
    for (int i = 0; i < 23; ++i) sum += __expf(F[i * 23 + j] - mx);
    const float inv = 1.0f / sum;
    for (int i = 0; i < 23; ++i)
      Af[i * 23 + j] = __expf(F[i * 23 + j] - mx) * inv
                     + A_ske[(h * 23 + i) * 23 + j] + attp[(h * 23 + i) * 23 + j];
  }
  __syncthreads();
  __bf16* dst = A_bf + (size_t)(b * 3 + h) * 32 * 32;
  for (int i = t; i < 32 * 32; i += 128) {
    const int k = i >> 5, j = i & 31;
    dst[i] = (k < 23 && j < 23) ? (__bf16)Af[k * 23 + j] : (__bf16)0.0f;
  }
}

// ---------------------------------------------------------------------------
// Kernel 4: per b:  gcn[b,o,m,l] = sum_h ( mw_h @ (X_l @ A_h) )[o,m] + sum_h mb_h[o]
// grid (256), block 128 (4 waves). Output bf16 [b][o][m(23)][l pad 64].
// ---------------------------------------------------------------------------
__global__ __launch_bounds__(128) void k_gcn(
    const float* __restrict__ x, const float* __restrict__ mw, const float* __restrict__ mb,
    const __bf16* __restrict__ A_bf, __bf16* __restrict__ gcn)
{
  __shared__ __bf16 MWs[3 * 64 * 64]; // [h][o][c]
  __shared__ __bf16 As[3 * 32 * 32];  // [h][k=n][m]
  __shared__ __bf16 Xs[64 * 32];      // [c][n]
  __shared__ __bf16 ST[32 * 64];      // [m][c]

  const int b = blockIdx.x, tid = threadIdx.x, w = tid >> 5, lane = tid & 31;
  const int hh = lane >> 4, col = lane & 15;

  for (int i = tid; i < 3 * 64 * 64; i += 128) MWs[i] = (__bf16)mw[i];
  const __bf16* abf = A_bf + (size_t)b * 3 * 32 * 32;
  for (int i = tid; i < 3 * 32 * 32; i += 128) As[i] = abf[i];

  float mbs[8];
#pragma unroll
  for (int r = 0; r < 8; ++r) {
    const int o = 16 * w + r + 8 * hh;
    mbs[r] = mb[o] + mb[64 + o] + mb[128 + o];
  }
  __syncthreads();

  const float* xb = x + (size_t)b * 64 * 23 * 35;
  for (int l = 0; l < 35; ++l) {
    for (int i = tid; i < 64 * 32; i += 128) {
      const int c = i >> 5, n = i & 31;
      Xs[i] = (n < 23) ? (__bf16)xb[(c * 23 + n) * 35 + l] : (__bf16)0.0f;
      if (n < 23 && l + 1 < 35) __builtin_prefetch(&xb[(c * 23 + n) * 35 + l + 1], 0, 0);
    }
    __syncthreads();

    const bf16x16 fx = load_frag_a(Xs + (16 * w) * 32, 32, lane);  // rows c, K=n
    floatx8 acc[2];
#pragma unroll
    for (int nt = 0; nt < 2; ++nt)
#pragma unroll
      for (int r = 0; r < 8; ++r) acc[nt][r] = mbs[r];

    for (int h = 0; h < 3; ++h) {
      // S = X_l @ A_h : wave w -> rows c in [16w, 16w+16)
      floatx8 s0 = {}, s1 = {};
      const bf16x16 fb0 = load_frag_b_strided(As + h * 1024 + 0, 32, lane);
      const bf16x16 fb1 = load_frag_b_strided(As + h * 1024 + 16, 32, lane);
      s0 = WMMA_BF16(fx, fb0, s0);
      s1 = WMMA_BF16(fx, fb1, s1);
#pragma unroll
      for (int r = 0; r < 8; ++r) {
        const int c = 16 * w + r + 8 * hh;
        ST[(0 + col) * 64 + c]  = (__bf16)s0[r];
        ST[(16 + col) * 64 + c] = (__bf16)s1[r];
      }
      __syncthreads();

      // ACC += MW_h @ S : rows o in [16w,16w+16), K = c = 64
#pragma unroll
      for (int kk = 0; kk < 2; ++kk) {
        const bf16x16 fm = load_frag_a(MWs + h * 4096 + (16 * w) * 64 + kk * 32, 64, lane);
#pragma unroll
        for (int nt = 0; nt < 2; ++nt) {
          const bf16x16 fs = load_frag_b_T(ST + (nt * 16) * 64 + kk * 32, 64, lane);
          acc[nt] = WMMA_BF16(fm, fs, acc[nt]);
        }
      }
      __syncthreads();  // before next h overwrites ST
    }

#pragma unroll
    for (int nt = 0; nt < 2; ++nt)
#pragma unroll
      for (int r = 0; r < 8; ++r) {
        const int o = 16 * w + r + 8 * hh, m = nt * 16 + col;
        if (m < 23) gcn[((size_t)(b * 64 + o) * 23 + m) * 64 + l] = (__bf16)acc[nt][r];
      }
  }
}

// ---------------------------------------------------------------------------
// Kernel 5: out[b,o,m,j] = sum_l gcn[b,o,m,l] * Wseq[l,j] + bias[j]   (f32 out)
// grid (23, 256), block 128 (4 waves); rows (o,m) tiled by 16 -> 92 tiles/b.
// ---------------------------------------------------------------------------
__global__ __launch_bounds__(128) void k_seq(
    const __bf16* __restrict__ gcn, const float* __restrict__ wseq,
    const float* __restrict__ bias, float* __restrict__ out)
{
  __shared__ __bf16 Wb[64 * 48];  // [l pad 64][j pad 48], zero padded
  __shared__ float  bs[48];

  const int tile = blockIdx.x, b = blockIdx.y;
  const int tid = threadIdx.x, w = tid >> 5, lane = tid & 31;
  const int hh = lane >> 4, col = lane & 15;

  for (int i = tid; i < 64 * 48; i += 128) {
    const int l = i / 48, j = i % 48;
    Wb[i] = (l < 35 && j < 35) ? (__bf16)wseq[l * 35 + j] : (__bf16)0.0f;
  }
  if (tid < 48) bs[tid] = (tid < 35) ? bias[tid] : 0.0f;
  __syncthreads();

  const int R = (tile * 4 + w) * 16;  // row base in [0,1472)
  const __bf16* ga = gcn + ((size_t)b * 1472 + R) * 64;

  floatx8 acc[3];
#pragma unroll
  for (int nt = 0; nt < 3; ++nt)
#pragma unroll
    for (int r = 0; r < 8; ++r) acc[nt][r] = bs[nt * 16 + col];

#pragma unroll
  for (int kk = 0; kk < 2; ++kk) {
    const bf16x16 fa = load_frag_a(ga + kk * 32, 64, lane);  // global bf16 rows
#pragma unroll
    for (int nt = 0; nt < 3; ++nt) {
      const bf16x16 fb = load_frag_b_strided(Wb + (kk * 32) * 48 + nt * 16, 48, lane);
      acc[nt] = WMMA_BF16(fa, fb, acc[nt]);
    }
  }

#pragma unroll
  for (int nt = 0; nt < 3; ++nt)
#pragma unroll
    for (int r = 0; r < 8; ++r) {
      const int row = R + r + 8 * hh, j = nt * 16 + col;
      if (j < 35) out[((size_t)b * 1472 + row) * 35 + j] = acc[nt][r];
    }
}

// ---------------------------------------------------------------------------
extern "C" void kernel_launch(void* const* d_in, const int* in_sizes, int n_in,
                              void* d_out, int out_size, void* d_ws, size_t ws_size,
                              hipStream_t stream) {
  const float* x    = (const float*)d_in[0];
  const float* cw1  = (const float*)d_in[1];
  const float* cb1  = (const float*)d_in[2];
  const float* cw2  = (const float*)d_in[3];
  const float* cb2  = (const float*)d_in[4];
  const float* bng  = (const float*)d_in[5];
  const float* bnb  = (const float*)d_in[6];
  const float* mw   = (const float*)d_in[7];
  const float* mb   = (const float*)d_in[8];
  const float* attp = (const float*)d_in[9];
  const float* aske = (const float*)d_in[10];
  const float* wseq = (const float*)d_in[11];
  const float* bias = (const float*)d_in[12];

  char* ws = (char*)d_ws;
  size_t off = 0;
  auto alloc = [&](size_t bytes) {
    size_t p = off;
    off = (off + bytes + 255) & ~(size_t)255;
    return p;
  };
  float*  att_ws   = (float*)(ws + alloc((size_t)256 * 3 * 529 * sizeof(float)));
  float*  scale_ws = (float*)(ws + alloc((size_t)3 * 529 * sizeof(float)));
  float*  shift_ws = (float*)(ws + alloc((size_t)3 * 529 * sizeof(float)));
  __bf16* A_bf     = (__bf16*)(ws + alloc((size_t)256 * 3 * 32 * 32 * sizeof(__bf16)));
  const size_t gcn_bytes = (size_t)256 * 1472 * 64 * sizeof(__bf16);
  __bf16* gcn_ws   = (__bf16*)(ws + alloc(gcn_bytes));

  // zero l-padding of the gcn staging buffer (graph-capture safe)
  hipMemsetAsync(gcn_ws, 0, gcn_bytes, stream);

  k_attm<<<dim3(256, 3), 128, 0, stream>>>(x, cw1, cb1, cw2, cb2, att_ws);
  k_bnstats<<<dim3(529, 3), 256, 0, stream>>>(att_ws, bng, bnb, scale_ws, shift_ws);
  k_softmax_A<<<dim3(256, 3), 128, 0, stream>>>(att_ws, scale_ws, shift_ws, attp, aske, A_bf);
  k_gcn<<<dim3(256), 128, 0, stream>>>(x, mw, mb, A_bf, gcn_ws);
  k_seq<<<dim3(23, 256), 128, 0, stream>>>(gcn_ws, wseq, bias, (float*)d_out);
}
